// CLGR_12171937317510
// MI455X (gfx1250) — matrix operand order repaired and checked by reading, not compiled, not run on hardware
//
#include <hip/hip_runtime.h>
#include <math.h>

typedef __attribute__((ext_vector_type(8)))  float  v8f;
typedef __attribute__((ext_vector_type(16))) __bf16 v16bf;

#define N_NODES 100000
#define N_EDGES 1600000
#define IN_DIM  512
#define HID_DIM 256
#define OUT_DIM 256

// ---------------- bf16 split helpers (round-to-nearest-even) ----------------
__device__ __forceinline__ unsigned short f2bf_rne(float f) {
    unsigned int u = __float_as_uint(f);
    unsigned int r = u + 0x7FFFu + ((u >> 16) & 1u);
    return (unsigned short)(r >> 16);
}
__device__ __forceinline__ float bf2f(unsigned short s) {
    return __uint_as_float(((unsigned int)s) << 16);
}
__device__ __forceinline__ __bf16 bfbits(unsigned short s) {
    union { unsigned short u; __bf16 b; } x; x.u = s; return x.b;
}

// ---------------- utility kernels ----------------
__global__ void fill_k(float* __restrict__ p, long long n, float v) {
    long long i = (long long)blockIdx.x * blockDim.x + threadIdx.x;
    long long stride = (long long)gridDim.x * blockDim.x;
    for (; i < n; i += stride) p[i] = v;
}

__global__ void deg_k(const int* __restrict__ dst, int E, float* __restrict__ deg) {
    int e = blockIdx.x * blockDim.x + threadIdx.x;
    if (e < E) atomicAdd(&deg[dst[e]], 1.0f);
}

__global__ void rsqrt_k(float* __restrict__ p, int n) {
    int i = blockIdx.x * blockDim.x + threadIdx.x;
    if (i < n) p[i] = rsqrtf(p[i]);
}

// Pre-layout weights into per-WMMA-fragment bf16 hi/lo order.
// B fragment k order used here (contiguous, same as A's): lane l covers column n = l&15,
// kbase = (l<16)?0:16, element j -> k = kbase + j  (2 bf16 per VGPR).
__global__ void wprep_k(const float* __restrict__ W, int K, int Nc,
                        __bf16* __restrict__ Whi, __bf16* __restrict__ Wlo) {
    int t = blockIdx.x * blockDim.x + threadIdx.x;
    int total = K * Nc;
    if (t >= total) return;
    int j    = t & 15;
    int l    = (t >> 4) & 31;
    int frag = t >> 9;                 // 512 elements per 32x16 fragment
    int NT   = Nc >> 4;
    int nt   = frag % NT;
    int kt   = frag / NT;
    int k = kt * 32 + ((l < 16) ? 0 : 16) + j;
    int n = nt * 16 + (l & 15);
    float w = W[(size_t)k * Nc + n];
    unsigned short hb = f2bf_rne(w);
    float rest = w - bf2f(hb);
    Whi[t] = bfbits(hb);
    Wlo[t] = bfbits(f2bf_rne(rest));
}

// ---------------- bf16x3 WMMA GEMM:  Out[M,Nc] = X[M,K] @ W[K,Nc] ----------------
// Block: 256 threads = 8 waves; block tile 64 rows x 32 cols; wave tile 16x16.
__global__ void __launch_bounds__(256)
gemm_bf16x3_k(const float* __restrict__ X, int M, int K, int Nc,
              const __bf16* __restrict__ Whi, const __bf16* __restrict__ Wlo,
              float* __restrict__ Out) {
    __shared__ float As[64][33];       // padded to kill LDS bank conflicts

    const int tid  = threadIdx.x;
    const int wave = tid >> 5;
    const int lane = tid & 31;
    const int rowTile = (wave & 3) * 16;    // 0,16,32,48
    const int colTile = (wave >> 2) * 16;   // 0,16
    const int blockRow = blockIdx.x * 64;
    const int blockCol = blockIdx.y * 32;
    const int NT = Nc >> 4;
    const int ntGlobal = (blockCol + colTile) >> 4;

    const int m      = lane & 15;
    const int kbaseA = (lane < 16) ? 0 : 16;

    v8f acc = {};

    for (int k0 = 0; k0 < K; k0 += 32) {
        // -------- cooperative fp32 A-tile load: 64x32 floats, 8 per thread --------
        {
            int idx = tid * 8;
            int r = idx >> 5;
            int c = idx & 31;
            float4 v0 = make_float4(0.f, 0.f, 0.f, 0.f);
            float4 v1 = make_float4(0.f, 0.f, 0.f, 0.f);
            if (blockRow + r < M) {
                const float4* gp = reinterpret_cast<const float4*>(
                    X + (size_t)(blockRow + r) * K + k0 + c);
                v0 = gp[0];
                v1 = gp[1];
            }
            As[r][c + 0] = v0.x; As[r][c + 1] = v0.y;
            As[r][c + 2] = v0.z; As[r][c + 3] = v0.w;
            As[r][c + 4] = v1.x; As[r][c + 5] = v1.y;
            As[r][c + 6] = v1.z; As[r][c + 7] = v1.w;
        }
        __syncthreads();

        // -------- build A fragments (16-bit A 16x32 layout), split hi/lo --------
        v16bf ah = {}, al = {};
#pragma unroll
        for (int e = 0; e < 16; ++e) {
            int k = kbaseA + e;   // {0..15} / {16..31}
            float a = As[rowTile + m][k];
            unsigned short hb = f2bf_rne(a);
            float rest = a - bf2f(hb);
            ah[e] = bfbits(hb);
            al[e] = bfbits(f2bf_rne(rest));
        }

        // -------- B fragments: contiguous 32B vector loads from pre-laid weights --------
        size_t fragBase = ((size_t)(k0 >> 5) * NT + ntGlobal) * 512 + (size_t)lane * 16;
        v16bf bh = *reinterpret_cast<const v16bf*>(Whi + fragBase);
        v16bf bl = *reinterpret_cast<const v16bf*>(Wlo + fragBase);

        // -------- bf16x3: ah*bh + ah*bl + al*bh, fp32 accumulate --------
        acc = __builtin_amdgcn_wmma_f32_16x16x32_bf16(false, ah, false, bh,
                                                      (short)0, acc, false, false);
        acc = __builtin_amdgcn_wmma_f32_16x16x32_bf16(false, ah, false, bl,
                                                      (short)0, acc, false, false);
        acc = __builtin_amdgcn_wmma_f32_16x16x32_bf16(false, al, false, bh,
                                                      (short)0, acc, false, false);
        __syncthreads();
    }

    // -------- store D (32-bit C/D layout: VGPR r -> M=r / M=r+8) --------
    const int col    = blockCol + colTile + (lane & 15);
    const int rowOff = blockRow + rowTile + ((lane >= 16) ? 8 : 0);
#pragma unroll
    for (int r = 0; r < 8; ++r) {
        int row = rowOff + r;
        if (row < M) Out[(size_t)row * Nc + col] = acc[r];
    }
}

// ---------------- edge aggregation: agg[dst] += dinv[src]*dinv[dst]*h[src] ----------------
// One block per edge; 256 threads = 256 features: coalesced 1KB gather,
// 1KB of atomics that stay resident in the 192MB L2 (h is 102MB).
__global__ void __launch_bounds__(256)
scatter_k(const float* __restrict__ h, const int* __restrict__ src,
          const int* __restrict__ dstv, const float* __restrict__ dinv,
          float* __restrict__ agg, int F) {
    int e = blockIdx.x;
    int f = threadIdx.x;
    int s = src[e];
    int d = dstv[e];
    float nrm = dinv[s] * dinv[d];
    atomicAdd(&agg[(size_t)d * F + f], nrm * h[(size_t)s * F + f]);
}

// out = agg + dinv^2 * h + bias  (optional relu)
__global__ void combine_k(const float* __restrict__ agg, const float* __restrict__ h,
                          const float* __restrict__ dinv, const float* __restrict__ bias,
                          long long total, int F, float* __restrict__ out, int relu) {
    long long idx = (long long)blockIdx.x * blockDim.x + threadIdx.x;
    if (idx >= total) return;
    long long i = idx / F;
    int f = (int)(idx - i * F);
    float dv = dinv[i];
    float v = agg[idx] + dv * dv * h[idx] + bias[f];
    if (relu) v = fmaxf(v, 0.0f);
    out[idx] = v;
}

// column sums / sums-of-squares: block reduces 256 rows locally, 2 atomics per column
__global__ void __launch_bounds__(256)
stats_k(const float* __restrict__ h, int M, int F, float* __restrict__ acc) {
    int f  = threadIdx.x;
    int r0 = blockIdx.x * 256;
    int r1 = r0 + 256; if (r1 > M) r1 = M;
    float s = 0.f, s2 = 0.f;
    for (int r = r0; r < r1; ++r) {
        float v = h[(size_t)r * F + f];
        s += v; s2 += v * v;
    }
    atomicAdd(&acc[f], s);
    atomicAdd(&acc[F + f], s2);
}

// z = (h - mean) / std  (ddof=1), in place
__global__ void norm_k(float* __restrict__ h, long long total, int F, int M,
                       const float* __restrict__ acc) {
    long long idx = (long long)blockIdx.x * blockDim.x + threadIdx.x;
    if (idx >= total) return;
    int f = (int)(idx % F);
    float s = acc[f], s2 = acc[F + f];
    float mean = s / (float)M;
    float var  = (s2 - s * s / (float)M) / (float)(M - 1);
    h[idx] = (h[idx] - mean) * rsqrtf(var);
}

// ---------------- host orchestration ----------------
extern "C" void kernel_launch(void* const* d_in, const int* in_sizes, int n_in,
                              void* d_out, int out_size, void* d_ws, size_t ws_size,
                              hipStream_t stream) {
    (void)in_sizes; (void)n_in; (void)out_size; (void)ws_size;

    const float* x[2]  = { (const float*)d_in[0], (const float*)d_in[1] };
    const int*   ei[2] = { (const int*)d_in[2],   (const int*)d_in[3]   };
    const float* W1 = (const float*)d_in[4];
    const float* b1 = (const float*)d_in[5];
    const float* W2 = (const float*)d_in[6];
    const float* b2 = (const float*)d_in[7];

    const int N = N_NODES, E = N_EDGES;
    const long long nf = (long long)N * HID_DIM;   // 25.6M elems per feature buffer

    // workspace carve-up
    float*  bufA  = (float*)d_ws;                  // GEMM output h
    float*  bufB  = bufA + nf;                     // agg1 / combined layer-1 activations
    float*  bufC  = bufB + nf;                     // agg2
    float*  dinv  = bufC + nf;                     // N floats (deg -> dinv in place)
    __bf16* W1hi  = (__bf16*)(dinv + N);
    __bf16* W1lo  = W1hi + (size_t)IN_DIM * HID_DIM;
    __bf16* W2hi  = W1lo + (size_t)IN_DIM * HID_DIM;
    __bf16* W2lo  = W2hi + (size_t)HID_DIM * OUT_DIM;
    float*  stats = (float*)(W2lo + (size_t)HID_DIM * OUT_DIM);   // 2*256 floats

    // one-time weight fragment prep (cheap; deterministic every call)
    wprep_k<<<(IN_DIM * HID_DIM + 255) / 256, 256, 0, stream>>>(W1, IN_DIM, HID_DIM, W1hi, W1lo);
    wprep_k<<<(HID_DIM * OUT_DIM + 255) / 256, 256, 0, stream>>>(W2, HID_DIM, OUT_DIM, W2hi, W2lo);

    const dim3 gemmGrid((N + 63) / 64, HID_DIM / 32);
    const int  combBlocks = (int)((nf + 255) / 256);

    for (int g = 0; g < 2; ++g) {
        const int* src  = ei[g];
        const int* dst  = ei[g] + E;
        float*     outg = (float*)d_out + (size_t)g * nf;

        // degrees (init 1.0 for self loop) -> dinv
        fill_k<<<1024, 256, 0, stream>>>(dinv, N, 1.0f);
        deg_k<<<(E + 255) / 256, 256, 0, stream>>>(dst, E, dinv);
        rsqrt_k<<<(N + 255) / 256, 256, 0, stream>>>(dinv, N);

        // ---- layer 1: P = X @ W1 (bf16x3 WMMA), aggregate, combine + relu ----
        gemm_bf16x3_k<<<gemmGrid, 256, 0, stream>>>(x[g], N, IN_DIM, HID_DIM, W1hi, W1lo, bufA);
        fill_k<<<2048, 256, 0, stream>>>(bufB, nf, 0.0f);
        scatter_k<<<E, 256, 0, stream>>>(bufA, src, dst, dinv, bufB, HID_DIM);
        combine_k<<<combBlocks, 256, 0, stream>>>(bufB, bufA, dinv, b1, nf, HID_DIM, bufB, 1);

        // ---- layer 2: P = H1 @ W2, aggregate, combine (no relu) -> d_out ----
        gemm_bf16x3_k<<<gemmGrid, 256, 0, stream>>>(bufB, N, HID_DIM, OUT_DIM, W2hi, W2lo, bufA);
        fill_k<<<2048, 256, 0, stream>>>(bufC, nf, 0.0f);
        scatter_k<<<E, 256, 0, stream>>>(bufA, src, dst, dinv, bufC, OUT_DIM);
        combine_k<<<combBlocks, 256, 0, stream>>>(bufC, bufA, dinv, b2, nf, OUT_DIM, outg, 0);

        // ---- per-column z-normalization (ddof=1), in place on d_out ----
        fill_k<<<2, 256, 0, stream>>>(stats, 2 * OUT_DIM, 0.0f);
        stats_k<<<(N + 255) / 256, 256, 0, stream>>>(outg, N, OUT_DIM, stats);
        norm_k<<<combBlocks, 256, 0, stream>>>(outg, nf, OUT_DIM, N, stats);
    }
}
